// GlobalPointer_70660801954221
// MI455X (gfx1250) — matrix operand order repaired and checked by reading, not compiled, not run on hardware
//
#include <hip/hip_runtime.h>
#include <math.h>

typedef __attribute__((ext_vector_type(16))) _Float16 v16h;
typedef __attribute__((ext_vector_type(8)))  _Float16 v8h;
typedef __attribute__((ext_vector_type(8)))  float    v8f;

#define HIDDEN    1024
#define HEADS     12
#define HEAD_SIZE 64
#define BATCH     8
#define SEQ       1024
#define QKF       (HEADS * HEAD_SIZE * 2)   /* 1536 projected features */
#define ROWS      (BATCH * SEQ)             /* 8192 flattened rows */

// swap value with lane^1 (ds_swizzle SWAPX1: xor=1, and=0x1f), wave32
__device__ __forceinline__ float pair_swap(float v) {
    return __int_as_float(__builtin_amdgcn_ds_swizzle(__float_as_int(v), 0x041F));
}

// Load a 16x32 f16 A-fragment from row-major src (lead dim ld):
// lane<16 holds row (lr), K {0..7, 16..23}; lane>=16 holds K {8..15, 24..31}.
__device__ __forceinline__ v16h load_afrag(const _Float16* __restrict__ src,
                                           int row, int ld, int k0, int lr, int hs) {
    const _Float16* p = src + (row + lr) * ld + k0 + hs * 8;
    v8h lo = *(const v8h*)(p);
    v8h hi = *(const v8h*)(p + 16);
    v16h a;
#pragma unroll
    for (int e = 0; e < 8; ++e) { a[e] = lo[e]; a[e + 8] = hi[e]; }
    return a;
}

// Load a 32x16 f16 B-fragment (B = srcRowMajor^T): lane n holds src row n;
// lane<16 K {0..7, 16..23}, lane>=16 K {8..15, 24..31}.
__device__ __forceinline__ v16h load_bfrag(const _Float16* __restrict__ src,
                                           int row, int ld, int k0, int lr, int hs) {
    const _Float16* p = src + (row + lr) * ld + k0 + hs * 8;
    v8h lo = *(const v8h*)(p);
    v8h hi = *(const v8h*)(p + 16);
    v16h b;
#pragma unroll
    for (int e = 0; e < 8; ++e) { b[e] = lo[e]; b[e + 8] = hi[e]; }
    return b;
}

// ---------------- f32 -> f16 bulk convert (X, W done once) -------------------
__global__ void cvt_f16_kernel(const float* __restrict__ src,
                               _Float16* __restrict__ dst, int n) {
    int idx = (blockIdx.x * blockDim.x + threadIdx.x) * 8;
    if (idx >= n) return;
    v8f v = *(const v8f*)(src + idx);
    v8h o;
#pragma unroll
    for (int e = 0; e < 8; ++e) o[e] = (_Float16)v[e];
    *(v8h*)(dst + idx) = o;
}

// ---------------- RoPE cos/sin tables: [pos 0..1023][i2 0..31] ----------------
__global__ void rope_table_kernel(float* __restrict__ ct, float* __restrict__ st) {
    int idx = blockIdx.x * blockDim.x + threadIdx.x;
    if (idx >= SEQ * 32) return;
    int pos = idx >> 5;
    int i2  = idx & 31;
    // inv_freq = 10000^(-(2*i2)/64) = exp(-(2*i2) * ln(10000)/64)
    float freq = expf(-(float)(2 * i2) * (9.210340371976184f / 64.0f));
    float ang  = (float)pos * freq;
    ct[idx] = cosf(ang);
    st[idx] = sinf(ang);
}

// ---------------- GEMM1: seq = X*W^T + b, fused RoPE, pack q/k to f16 --------
// wave tile 64x64, 4x4 register block of v_wmma_f32_16x16x32_f16.
__global__ __launch_bounds__(128) void qkproj_rope_kernel(
    const _Float16* __restrict__ X16,  // (8192, 1024) f16
    const _Float16* __restrict__ W16,  // (1536, 1024) f16
    const float* __restrict__ bias,    // (1536)
    const float* __restrict__ ct, const float* __restrict__ st,
    _Float16* __restrict__ qws,        // (8,12,1024,64)
    _Float16* __restrict__ kws)        // (8,12,1024,64)
{
    const int lane = threadIdx.x;          // 0..31
    const int wy   = threadIdx.y;          // 0..3 (wave id)
    const int m0 = (blockIdx.y * 4 + wy) * 64;
    const int n0 = blockIdx.x * 64;
    const int lr = lane & 15;
    const int hs = lane >> 4;              // half-wave select

    v8f acc[4][4];
#pragma unroll
    for (int i = 0; i < 4; ++i)
#pragma unroll
        for (int j = 0; j < 4; ++j)
#pragma unroll
            for (int e = 0; e < 8; ++e) acc[i][j][e] = 0.0f;

#pragma unroll 2
    for (int k0 = 0; k0 < HIDDEN; k0 += 32) {
        v16h afr[4], bfr[4];
#pragma unroll
        for (int mi = 0; mi < 4; ++mi)
            afr[mi] = load_afrag(X16, m0 + mi * 16, HIDDEN, k0, lr, hs);
#pragma unroll
        for (int ni = 0; ni < 4; ++ni)
            bfr[ni] = load_bfrag(W16, n0 + ni * 16, HIDDEN, k0, lr, hs);
#pragma unroll
        for (int mi = 0; mi < 4; ++mi)
#pragma unroll
            for (int ni = 0; ni < 4; ++ni)
                acc[mi][ni] = __builtin_amdgcn_wmma_f32_16x16x32_f16(
                    false, afr[mi], false, bfr[ni], (short)0, acc[mi][ni], false, false);
    }

    // Epilogue: + bias, RoPE (partner via ds_swizzle xor1), store f16 q/k
#pragma unroll
    for (int ni = 0; ni < 4; ++ni) {
        const int c    = n0 + ni * 16 + lr;       // feature 0..1535
        const int h    = c >> 7;
        const int r128 = c & 127;
        const bool isq = r128 < HEAD_SIZE;
        const int d    = isq ? r128 : (r128 - HEAD_SIZE);
        const int i2   = d >> 1;
        const float cb  = bias[c];
        const float sgn = (d & 1) ? 1.0f : -1.0f; // even: -partner*sin, odd: +partner*sin
        _Float16* dst = isq ? qws : kws;
#pragma unroll
        for (int mi = 0; mi < 4; ++mi) {
#pragma unroll
            for (int r = 0; r < 8; ++r) {
                const int row = m0 + mi * 16 + hs * 8 + r;  // 0..8191
                const int pos = row & (SEQ - 1);
                const int bi  = row >> 10;
                float v  = acc[mi][ni][r] + cb;
                float pv = pair_swap(v);
                const float cs = ct[pos * 32 + i2];
                const float sn = st[pos * 32 + i2];
                float o = v * cs + sgn * pv * sn;
                dst[(((size_t)(bi * HEADS + h) * SEQ) + pos) * HEAD_SIZE + d] = (_Float16)o;
            }
        }
    }
}

// ---------------- GEMM2: logits[b,h] = Q K^T, mask + causal + scale ----------
__global__ __launch_bounds__(128) void logits_kernel(
    const _Float16* __restrict__ qws, const _Float16* __restrict__ kws,
    const int* __restrict__ mask, float* __restrict__ out)
{
    const int lane = threadIdx.x;
    const int wy   = threadIdx.y;
    const int bh   = blockIdx.z;           // b*HEADS + h
    const int b    = bh / HEADS;
    const int m0 = (blockIdx.y * 4 + wy) * 64;
    const int n0 = blockIdx.x * 64;
    const int lr = lane & 15;
    const int hs = lane >> 4;

    const _Float16* Q = qws + (size_t)bh * SEQ * HEAD_SIZE;
    const _Float16* K = kws + (size_t)bh * SEQ * HEAD_SIZE;

    v8f acc[4][4];
#pragma unroll
    for (int i = 0; i < 4; ++i)
#pragma unroll
        for (int j = 0; j < 4; ++j)
#pragma unroll
            for (int e = 0; e < 8; ++e) acc[i][j][e] = 0.0f;

#pragma unroll
    for (int k0 = 0; k0 < HEAD_SIZE; k0 += 32) {
        v16h afr[4], bfr[4];
#pragma unroll
        for (int mi = 0; mi < 4; ++mi)
            afr[mi] = load_afrag(Q, m0 + mi * 16, HEAD_SIZE, k0, lr, hs);
#pragma unroll
        for (int ni = 0; ni < 4; ++ni)
            bfr[ni] = load_bfrag(K, n0 + ni * 16, HEAD_SIZE, k0, lr, hs);
#pragma unroll
        for (int mi = 0; mi < 4; ++mi)
#pragma unroll
            for (int ni = 0; ni < 4; ++ni)
                acc[mi][ni] = __builtin_amdgcn_wmma_f32_16x16x32_f16(
                    false, afr[mi], false, bfr[ni], (short)0, acc[mi][ni], false, false);
    }

    const int* bm = mask + b * SEQ;
#pragma unroll
    for (int ni = 0; ni < 4; ++ni) {
        const int gn = n0 + ni * 16 + lr;
        const int nm = bm[gn];
#pragma unroll
        for (int mi = 0; mi < 4; ++mi) {
#pragma unroll
            for (int r = 0; r < 8; ++r) {
                const int gm = m0 + mi * 16 + hs * 8 + r;
                const int mm = bm[gm];
                float v;
                if (mm && nm) {
                    v = acc[mi][ni][r];
                    if (gm > gn) v -= 1.0e12f;   // tril(-1) causal bias
                    v *= 0.125f;                  // 1/sqrt(64)
                } else {
                    v = -INFINITY;
                }
                out[((size_t)bh * SEQ + gm) * SEQ + gn] = v;
            }
        }
    }
}

extern "C" void kernel_launch(void* const* d_in, const int* in_sizes, int n_in,
                              void* d_out, int out_size, void* d_ws, size_t ws_size,
                              hipStream_t stream) {
    const float* X    = (const float*)d_in[0];
    const int*   mask = (const int*)d_in[1];
    const float* W    = (const float*)d_in[2];
    const float* bias = (const float*)d_in[3];
    float* out = (float*)d_out;

    const size_t NX = (size_t)ROWS * HIDDEN;                    // 8,388,608
    const size_t NW = (size_t)QKF * HIDDEN;                     // 1,572,864
    const size_t QH = (size_t)BATCH * HEADS * SEQ * HEAD_SIZE;  // 6,291,456

    _Float16* x16 = (_Float16*)d_ws;
    _Float16* w16 = x16 + NX;
    _Float16* qws = w16 + NW;
    _Float16* kws = qws + QH;
    float* ct = (float*)(kws + QH);
    float* st = ct + SEQ * 32;

    cvt_f16_kernel<<<(int)(NX / 8 + 255) / 256, 256, 0, stream>>>(X, x16, (int)NX);
    cvt_f16_kernel<<<(int)(NW / 8 + 255) / 256, 256, 0, stream>>>(W, w16, (int)NW);
    rope_table_kernel<<<(SEQ * 32 + 255) / 256, 256, 0, stream>>>(ct, st);

    qkproj_rope_kernel<<<dim3(QKF / 64, ROWS / 256), dim3(32, 4), 0, stream>>>(
        x16, w16, bias, ct, st, qws, kws);

    logits_kernel<<<dim3(SEQ / 64, SEQ / 256, BATCH * HEADS), dim3(32, 4), 0, stream>>>(
        qws, kws, mask, out);
}
